// TwistorLNN_25125558682290
// MI455X (gfx1250) — matrix operand order repaired and checked by reading, not compiled, not run on hardware
//
#include <hip/hip_runtime.h>
#include <hip/hip_bf16.h>

// TwistorLNN persistent-scan kernel for gfx1250 (MI455X).
// - Weights register-resident as WMMA B-fragments (fp16), zero weight traffic in t-loop
// - State (zr, zi) resident in LDS per workgroup (16 batch rows / WG, 32 WGs)
// - All four GEMMs per step via v_wmma_f32_16x16x32_f16
// - Per-step device-wide mean-norm via per-t accumulator slots + monotonic counter

#define T_STEPS  512
#define BATCH    512
#define IN_DIM   64
#define H_DIM    128
#define NWG      32
#define ROWS     16          // batch rows per workgroup (NWG*ROWS == BATCH)
#define NTHREADS 256         // 8 wave32s

typedef _Float16 v16h __attribute__((ext_vector_type(16)));
typedef _Float16 v8h  __attribute__((ext_vector_type(8)));
typedef float    v8f  __attribute__((ext_vector_type(8)));

__device__ __forceinline__ float sigmoidf_fast(float xv) {
    return 1.0f / (1.0f + __expf(-xv));
}

// A-fragment (16x32 fp16) from LDS, row-major, rowstride rs halfs.
// Lane layout: lanes 0-15 hold row m=lane, K = k0+[0..15];
// lanes 16-31 hold row m=lane-16, K = k0+[16..31].
__device__ __forceinline__ v16h load_a_frag(const _Float16* base, int lane, int rs, int k0) {
    const int m  = lane & 15;
    const int kb = k0 + ((lane & 16) ? 16 : 0);
    const _Float16* p = base + m * rs + kb;
    v8h lo = *(const v8h*)(p);
    v8h hi = *(const v8h*)(p + 8);
    return __builtin_shufflevector(lo, hi, 0,1,2,3,4,5,6,7,8,9,10,11,12,13,14,15);
}

// B-fragment (32x16 fp16) for D = A x B with B[k][n] = W[n][k]:
// lane (n = lane&15) needs a contiguous 16-half run of row n of W -> direct from global fp32,
// optionally applying the learned-sparsity gate sigmoid(mask).
__device__ __forceinline__ v16h load_b_eff(const float* W, const float* Msk,
                                           int lane, int rs, int n0, int k0) {
    const int n  = n0 + (lane & 15);
    const int kb = k0 + ((lane & 16) ? 16 : 0);
    const float* wp = W + n * rs + kb;
    const float* mp = Msk ? (Msk + n * rs + kb) : nullptr;
    v16h r;
#pragma unroll
    for (int j = 0; j < 16; ++j) {
        float w = wp[j];
        if (mp) w *= sigmoidf_fast(mp[j]);
        r[j] = (_Float16)w;
    }
    return r;
}

__global__ void zero_ws_kernel(float* sums, unsigned* counter) {
    int i = threadIdx.x + blockIdx.x * blockDim.x;
    if (i < T_STEPS) sums[i] = 0.0f;
    if (i == 0) *counter = 0u;
}

__global__ __launch_bounds__(NTHREADS) void twistor_lnn_persistent(
    const float* __restrict__ x,
    const float* __restrict__ Wr,  const float* __restrict__ Wrb,
    const float* __restrict__ Wi,  const float* __restrict__ Wib,
    const float* __restrict__ Uw,  const float* __restrict__ Ub,
    const float* __restrict__ Wt,  const float* __restrict__ Wtb,
    const float* __restrict__ Mr,  const float* __restrict__ Mi,
    const float* __restrict__ taub,
    const float* __restrict__ br,  const float* __restrict__ bi,
    const float* __restrict__ ow,  const float* __restrict__ ob,
    float* __restrict__ sums, unsigned* __restrict__ counter,
    float* __restrict__ out)
{
    __shared__ float    sZr[ROWS * H_DIM];
    __shared__ float    sZi[ROWS * H_DIM];
    __shared__ _Float16 sTr[ROWS * H_DIM];
    __shared__ _Float16 sTi[ROWS * H_DIM];
    __shared__ _Float16 sZm[ROWS * H_DIM];
    __shared__ _Float16 sX [ROWS * IN_DIM];
    __shared__ float    sOw[H_DIM];
    __shared__ float    sRed[NTHREADS / 32];
    __shared__ float    sScale;

    const int tid  = threadIdx.x;
    const int lane = tid & 31;
    const int wv   = tid >> 5;          // wave id 0..7
    const int b0   = blockIdx.x * ROWS; // batch-row base for this WG
    const int n0   = wv * 16;           // output-column base for this wave
    const int nn   = n0 + (lane & 15);  // this lane's n (C/D layout)

    // -------- register-resident effective weights (B fragments, fp16) --------
    v16h bWr[4], bWi[4], bWt[4], bU[2];
#pragma unroll
    for (int kk = 0; kk < 4; ++kk) {
        bWr[kk] = load_b_eff(Wr, Mr,      lane, H_DIM, n0, kk * 32);
        bWi[kk] = load_b_eff(Wi, Mi,      lane, H_DIM, n0, kk * 32);
        bWt[kk] = load_b_eff(Wt, nullptr, lane, H_DIM, n0, kk * 32);
    }
#pragma unroll
    for (int kk = 0; kk < 2; ++kk)
        bU[kk] = load_b_eff(Uw, nullptr, lane, IN_DIM, n0, kk * 32);

    const float biasR = Wrb[nn] + Ub[nn] + br[nn];
    const float biasI = Wib[nn] + Ub[nn] + bi[nn];
    const float biasT = Wtb[nn];
    const float tbias = taub[nn];
    const float outb  = ob[0];

    for (int i = tid; i < H_DIM; i += NTHREADS) sOw[i] = ow[i];
    for (int i = tid; i < ROWS * H_DIM; i += NTHREADS) { sZr[i] = 0.0f; sZi[i] = 0.0f; }
    __syncthreads();

    const int mofs = (lane & 16) ? 8 : 0; // C/D layout: vgpr j -> m = j + mofs

    for (int t = 0; t < T_STEPS; ++t) {
        // ---- stage: tanh(z), |z|, x(t) -> fp16 LDS ----
        for (int i = tid; i < ROWS * H_DIM; i += NTHREADS) {
            float zr = sZr[i], zi = sZi[i];
            sTr[i] = (_Float16)tanhf(zr);
            sTi[i] = (_Float16)tanhf(zi);
            sZm[i] = (_Float16)sqrtf(zr * zr + zi * zi);
        }
        const float* xt = x + ((size_t)t * BATCH + b0) * IN_DIM;
        for (int i = tid; i < ROWS * IN_DIM; i += NTHREADS)
            sX[i] = (_Float16)xt[i];
        if (t + 1 < T_STEPS)
            __builtin_prefetch(x + ((size_t)(t + 1) * BATCH + b0) * IN_DIM + tid, 0, 0);
        __syncthreads();

        // ---- WMMA GEMMs: one 16x16 output tile per wave, uniform EXEC ----
        v8f accR = {}, accI = {}, accT = {}, accU = {};
#pragma unroll
        for (int kk = 0; kk < 4; ++kk) {
            v16h aR = load_a_frag(sTr, lane, H_DIM, kk * 32);
            accR = __builtin_amdgcn_wmma_f32_16x16x32_f16(false, aR, false, bWr[kk],
                                                          (short)0, accR, false, false);
            v16h aI = load_a_frag(sTi, lane, H_DIM, kk * 32);
            accI = __builtin_amdgcn_wmma_f32_16x16x32_f16(false, aI, false, bWi[kk],
                                                          (short)0, accI, false, false);
            v16h aM = load_a_frag(sZm, lane, H_DIM, kk * 32);
            accT = __builtin_amdgcn_wmma_f32_16x16x32_f16(false, aM, false, bWt[kk],
                                                          (short)0, accT, false, false);
        }
#pragma unroll
        for (int kk = 0; kk < 2; ++kk) {
            v16h aX = load_a_frag(sX, lane, IN_DIM, kk * 32);
            accU = __builtin_amdgcn_wmma_f32_16x16x32_f16(false, aX, false, bU[kk],
                                                          (short)0, accU, false, false);
        }

        // ---- elementwise dynamics + local norm partial sum ----
        float dR[8], dI[8];
        float lsum = 0.0f;
#pragma unroll
        for (int j = 0; j < 8; ++j) {
            const int idx = (j + mofs) * H_DIM + nn;
            float zr = sZr[idx], zi = sZi[idx];
            float ux = accU[j];
            float dzr = -zr + accR[j] + ux + biasR;
            float dzi = -zi + accI[j] + ux + biasI;
            float tau = sigmoidf_fast(accT[j] + biasT) + tbias;
            tau = fminf(fmaxf(tau, 0.01f), 1.0f) + 1e-6f;
            dzr = fminf(fmaxf(dzr / tau, -10.0f), 10.0f);
            dzi = fminf(fmaxf(dzi / tau, -10.0f), 10.0f);
            lsum += sqrtf(dzr * dzr + dzi * dzi);
            dR[j] = dzr; dI[j] = dzi;
        }
#pragma unroll
        for (int off = 16; off > 0; off >>= 1)
            lsum += __shfl_xor(lsum, off, 32);
        if (lane == 0) sRed[wv] = lsum;
        __syncthreads();

        // ---- device-wide mean-norm: per-t slot + monotonic arrival counter ----
        if (tid == 0) {
            float wg = 0.0f;
#pragma unroll
            for (int i = 0; i < NTHREADS / 32; ++i) wg += sRed[i];
            atomicAdd(&sums[t], wg);
            __threadfence();
            atomicAdd(counter, 1u);
            const unsigned target = (unsigned)NWG * (unsigned)(t + 1);
            while (__hip_atomic_load(counter, __ATOMIC_ACQUIRE,
                                     __HIP_MEMORY_SCOPE_AGENT) < target)
                __builtin_amdgcn_s_sleep(1);
            float tot  = __hip_atomic_load(&sums[t], __ATOMIC_RELAXED,
                                           __HIP_MEMORY_SCOPE_AGENT);
            float mean = tot * (1.0f / (float)(BATCH * H_DIM));
            sScale = (mean > 5.0f) ? (5.0f / (mean + 1e-6f)) : 1.0f;
        }
        __syncthreads();

        // ---- state update (each lane owns its (m,n) elements) ----
        const float sc = sScale;
#pragma unroll
        for (int j = 0; j < 8; ++j) {
            const int idx = (j + mofs) * H_DIM + nn;
            float zr = sZr[idx] + 0.1f * dR[j] * sc;
            float zi = sZi[idx] + 0.1f * dI[j] * sc;
            sZr[idx] = fminf(fmaxf(zr, -100.0f), 100.0f);
            sZi[idx] = fminf(fmaxf(zi, -100.0f), 100.0f);
        }
        __syncthreads();

        // ---- readout: out[t, b] = dot(zr[b,:], ow) + ob ; 8 lanes per row ----
        if (tid < ROWS * 8) {
            const int row = tid >> 3;
            const int seg = tid & 7;
            float p = 0.0f;
#pragma unroll
            for (int h = 0; h < 16; ++h)
                p += sZr[row * H_DIM + seg * 16 + h] * sOw[seg * 16 + h];
            p += __shfl_xor(p, 1, 32);
            p += __shfl_xor(p, 2, 32);
            p += __shfl_xor(p, 4, 32);
            if (seg == 0)
                out[(size_t)t * BATCH + b0 + row] = p + outb;
        }
        __syncthreads();
    }
}

extern "C" void kernel_launch(void* const* d_in, const int* in_sizes, int n_in,
                              void* d_out, int out_size, void* d_ws, size_t ws_size,
                              hipStream_t stream) {
    (void)in_sizes; (void)n_in; (void)out_size; (void)ws_size;
    const float* x    = (const float*)d_in[0];
    const float* Wr   = (const float*)d_in[1];
    const float* Wrb  = (const float*)d_in[2];
    const float* Wi   = (const float*)d_in[3];
    const float* Wib  = (const float*)d_in[4];
    const float* Uw   = (const float*)d_in[5];
    const float* Ub   = (const float*)d_in[6];
    const float* Wt   = (const float*)d_in[7];
    const float* Wtb  = (const float*)d_in[8];
    const float* Mr   = (const float*)d_in[9];
    const float* Mi   = (const float*)d_in[10];
    const float* taub = (const float*)d_in[11];
    const float* br   = (const float*)d_in[12];
    const float* bi   = (const float*)d_in[13];
    const float* ow   = (const float*)d_in[14];
    const float* ob   = (const float*)d_in[15];

    float*    sums    = (float*)d_ws;
    unsigned* counter = (unsigned*)((char*)d_ws + T_STEPS * sizeof(float));

    zero_ws_kernel<<<2, 256, 0, stream>>>(sums, counter);
    twistor_lnn_persistent<<<NWG, NTHREADS, 0, stream>>>(
        x, Wr, Wrb, Wi, Wib, Uw, Ub, Wt, Wtb, Mr, Mi, taub, br, bi, ow, ob,
        sums, counter, (float*)d_out);
}
